// attn_v2_57286273794065
// MI455X (gfx1250) — matrix operand order repaired and checked by reading, not compiled, not run on hardware
//
#include <hip/hip_runtime.h>
#include <cstdint>
#include <cstddef>

#define B_  4
#define C_  256
#define W_  128
#define HW_ 16384
#define N2_ 4096

typedef __attribute__((ext_vector_type(16))) __bf16 v16bf;
typedef __attribute__((ext_vector_type(8)))  float  v8f;

union BfOp { uint4 u[2]; v16bf v; };

__device__ __forceinline__ unsigned short f2bf(float f) {
  unsigned int u = __float_as_uint(f);
  unsigned int r = u + 0x7FFFu + ((u >> 16) & 1u);   // round-to-nearest-even
  return (unsigned short)(r >> 16);
}

// ---------------------------------------------------------------- K1: ori = t+b+m, per-(b,c) mean/max of t,b,m
__global__ void k1_stats_ori(const float* __restrict__ t, const float* __restrict__ bsrc,
                             const float* __restrict__ m, float* __restrict__ ori,
                             float* __restrict__ sMean, float* __restrict__ sMax) {
  int bc = blockIdx.x, tid = threadIdx.x;
  size_t base = (size_t)bc * HW_;
  float s0 = 0.f, s1 = 0.f, s2 = 0.f;
  float m0 = -1e30f, m1 = -1e30f, m2 = -1e30f;
  for (int i = tid * 4; i < HW_; i += 1024) {
    float4 a = *(const float4*)(t + base + i);
    float4 b = *(const float4*)(bsrc + base + i);
    float4 c = *(const float4*)(m + base + i);
    float4 o;
    o.x = a.x + b.x + c.x; o.y = a.y + b.y + c.y;
    o.z = a.z + b.z + c.z; o.w = a.w + b.w + c.w;
    *(float4*)(ori + base + i) = o;
    s0 += a.x + a.y + a.z + a.w; m0 = fmaxf(m0, fmaxf(fmaxf(a.x, a.y), fmaxf(a.z, a.w)));
    s1 += b.x + b.y + b.z + b.w; m1 = fmaxf(m1, fmaxf(fmaxf(b.x, b.y), fmaxf(b.z, b.w)));
    s2 += c.x + c.y + c.z + c.w; m2 = fmaxf(m2, fmaxf(fmaxf(c.x, c.y), fmaxf(c.z, c.w)));
  }
  __shared__ float r[6][256];
  r[0][tid] = s0; r[1][tid] = s1; r[2][tid] = s2;
  r[3][tid] = m0; r[4][tid] = m1; r[5][tid] = m2;
  for (int off = 128; off > 0; off >>= 1) {
    __syncthreads();
    if (tid < off) {
      r[0][tid] += r[0][tid + off]; r[1][tid] += r[1][tid + off]; r[2][tid] += r[2][tid + off];
      r[3][tid] = fmaxf(r[3][tid], r[3][tid + off]);
      r[4][tid] = fmaxf(r[4][tid], r[4][tid + off]);
      r[5][tid] = fmaxf(r[5][tid], r[5][tid + off]);
    }
  }
  __syncthreads();
  if (tid == 0) {
    const float inv = 1.0f / (float)HW_;
    sMean[0 * 1024 + bc] = r[0][0] * inv;
    sMean[1 * 1024 + bc] = r[1][0] * inv;
    sMean[2 * 1024 + bc] = r[2][0] * inv;
    sMax[0 * 1024 + bc] = r[3][0];
    sMax[1 * 1024 + bc] = r[4][0];
    sMax[2 * 1024 + bc] = r[5][0];
  }
}

// ---------------------------------------------------------------- K2: channel attention -> addC[b*C+c] = gamma_c * v
__global__ void k2_chan_attn(const float* __restrict__ sMean, const float* __restrict__ sMax,
                             const float* __restrict__ ckw1, const float* __restrict__ ckw2,
                             const float* __restrict__ cqw1, const float* __restrict__ cqw2,
                             const float* __restrict__ cvw1, const float* __restrict__ cvw2,
                             const float* __restrict__ gammac, float* __restrict__ addC) {
  int b = blockIdx.x, tid = threadIdx.x;
  __shared__ float hid[3][2][16];
  __shared__ float fA[256], gA[256], hA[256];
  __shared__ float sred[2];
  if (tid < 96) {
    int tensor = tid / 32, rem = tid & 31, stat = rem >> 4, j = rem & 15;
    const float* w1 = (tensor == 0) ? ckw1 : ((tensor == 1) ? cqw1 : cvw1);
    const float* st = (stat ? sMax : sMean) + tensor * 1024 + b * 256;
    float s = 0.f;
    for (int c = 0; c < 256; ++c) s += st[c] * w1[j * 256 + c];
    hid[tensor][stat][j] = fmaxf(s, 0.f);
  }
  __syncthreads();
  {
    int o = tid;
    float sf = 0.f, sg = 0.f, sh = 0.f;
    for (int j = 0; j < 16; ++j) {
      sf += (hid[0][0][j] + hid[0][1][j]) * ckw2[o * 16 + j];
      sg += (hid[1][0][j] + hid[1][1][j]) * cqw2[o * 16 + j];
      sh += (hid[2][0][j] + hid[2][1][j]) * cvw2[o * 16 + j];
    }
    fA[o] = sf; gA[o] = sg; hA[o] = sh;
  }
  __syncthreads();
  if (tid == 0) {
    float mx = -1e30f, mn = 1e30f;
    for (int i = 0; i < 256; ++i) { mx = fmaxf(mx, fA[i]); mn = fminf(mn, fA[i]); }
    sred[0] = mx; sred[1] = mn;
  }
  __syncthreads();
  float fmx = sred[0], fmn = sred[1];
  float gj = gA[tid];
  float mj = (gj >= 0.f) ? gj * fmx : gj * fmn;
  float num = 0.f, den = 0.f;
  for (int i = 0; i < 256; ++i) {
    float e = __expf(fA[i] * gj - mj);
    den += e; num += hA[i] * e;
  }
  addC[b * 256 + tid] = gammac[0] * num / den;
}

// ---------------------------------------------------------------- K4: weights -> bf16 (nt,nb,nm and sum)
__global__ void k4_wprep(const float* __restrict__ ntw, const float* __restrict__ nbw,
                         const float* __restrict__ nmw, unsigned short* __restrict__ W3,
                         unsigned short* __restrict__ Wsum) {
  int idx = blockIdx.x * 256 + threadIdx.x;   // 65536
  float a = ntw[idx], b = nbw[idx], c = nmw[idx];
  W3[idx] = f2bf(a);
  W3[65536 + idx] = f2bf(b);
  W3[131072 + idx] = f2bf(c);
  Wsum[idx] = f2bf(a + b + c);
}

// ---------------------------------------------------------------- K3: X = bf16(ori + addC), Xp = 2x2 avg pool (bf16)
__global__ void k3_fuse_pool(const float* __restrict__ ori, const float* __restrict__ addC,
                             unsigned short* __restrict__ X, unsigned short* __restrict__ Xp) {
  int idx = blockIdx.x * 256 + threadIdx.x;   // B*C*N2
  int p2 = idx & (N2_ - 1);
  int bc = idx >> 12;
  int y2 = p2 >> 6, x2 = p2 & 63;
  size_t base = (size_t)bc * HW_ + (size_t)(y2 * 2) * W_ + (size_t)(x2 * 2);
  float a = addC[bc];
  float o00 = ori[base] + a, o01 = ori[base + 1] + a;
  float o10 = ori[base + W_] + a, o11 = ori[base + W_ + 1] + a;
  X[base] = f2bf(o00); X[base + 1] = f2bf(o01);
  X[base + W_] = f2bf(o10); X[base + W_ + 1] = f2bf(o11);
  Xp[(size_t)bc * N2_ + p2] = f2bf(0.25f * (o00 + o01 + o10 + o11));
}

// ---------------------------------------------------------------- WMMA GEMM core: D[256 x 64cols] = W[256x256] * X[256 x Npix]
// 8 waves: waveM in 0..3 (64 rows each), waveN in 0..1 (32 cols each).
// A (16x32 bf16) per ISA: lane m=l&15, elems i<8 -> K=8*half+i ; i>=8 -> K=16+8*half+(i-8)
// B (32x16 bf16): lane n=l&15, elems i<8 -> K=8*half+i ; i>=8 -> K=16+8*half+(i-8)  (staged transposed in LDS)
__device__ __forceinline__ void gemm_core(const unsigned short* __restrict__ Xb, int Npix, int pixBase,
                                          const unsigned short* __restrict__ Wb,
                                          unsigned short (*XT)[40], v8f acc[4][2]) {
  int tid = threadIdx.x;
  int lane = tid & 31, wave = tid >> 5;
  int waveM = wave >> 1, waveN = wave & 1;
  int half = lane >> 4, l15 = lane & 15;
  int lk = tid >> 3;            // staging: k row 0..31
  int lc = (tid & 7) * 8;       // staging: col block
  for (int kk = 0; kk < 8; ++kk) {
    int K0 = kk * 32;
    __syncthreads();
    {
      uint4 d = *reinterpret_cast<const uint4*>(Xb + (size_t)(K0 + lk) * Npix + pixBase + lc);
      union { uint4 u; unsigned short s[8]; } w;
      w.u = d;
#pragma unroll
      for (int i = 0; i < 8; ++i) XT[lc + i][lk] = w.s[i];
    }
    __syncthreads();
    BfOp Aop[4];
#pragma unroll
    for (int mt = 0; mt < 4; ++mt) {
      const unsigned short* wr = Wb + (size_t)(waveM * 64 + mt * 16 + l15) * 256 + K0;
      Aop[mt].u[0] = *reinterpret_cast<const uint4*>(wr + 8 * half);
      Aop[mt].u[1] = *reinterpret_cast<const uint4*>(wr + 16 + 8 * half);
    }
    BfOp Bop[2];
#pragma unroll
    for (int nt = 0; nt < 2; ++nt) {
      const unsigned short* bp = &XT[waveN * 32 + nt * 16 + l15][0];
      Bop[nt].u[0] = *reinterpret_cast<const uint4*>(bp + 8 * half);
      Bop[nt].u[1] = *reinterpret_cast<const uint4*>(bp + 16 + 8 * half);
    }
#pragma unroll
    for (int mt = 0; mt < 4; ++mt)
#pragma unroll
      for (int nt = 0; nt < 2; ++nt)
        acc[mt][nt] = __builtin_amdgcn_wmma_f32_16x16x32_bf16(
            false, Aop[mt].v, false, Bop[nt].v, (short)0, acc[mt][nt], false, false);
  }
}

// ---------------------------------------------------------------- K5: full-res ori_s = Wsum @ X
__global__ void k5_gemm_full(const unsigned short* __restrict__ X, const unsigned short* __restrict__ Wsum,
                             float* __restrict__ oriS) {
  __shared__ __align__(16) unsigned short XT[64][40];
  int b = blockIdx.z;
  int pixBase = blockIdx.x * 64;
  v8f acc[4][2];
#pragma unroll
  for (int i = 0; i < 4; ++i)
#pragma unroll
    for (int j = 0; j < 2; ++j)
#pragma unroll
      for (int e = 0; e < 8; ++e) acc[i][j][e] = 0.f;
  gemm_core(X + (size_t)b * C_ * HW_, HW_, pixBase, Wsum, XT, acc);
  int lane = threadIdx.x & 31, wave = threadIdx.x >> 5;
  int waveM = wave >> 1, waveN = wave & 1, half = lane >> 4, l15 = lane & 15;
  float* ob = oriS + (size_t)b * C_ * HW_;
#pragma unroll
  for (int mt = 0; mt < 4; ++mt)
#pragma unroll
    for (int nt = 0; nt < 2; ++nt) {
      int n = pixBase + waveN * 32 + nt * 16 + l15;
#pragma unroll
      for (int v = 0; v < 8; ++v) {
        int row = waveM * 64 + mt * 16 + v + 8 * half;
        ob[(size_t)row * HW_ + n] = acc[mt][nt][v];
      }
    }
}

// ---------------------------------------------------------------- K6: pooled GEMM + channel mean/max reduction
__global__ void k6_gemm_pool(const unsigned short* __restrict__ Xp, const unsigned short* __restrict__ W3,
                             float* __restrict__ meanMaps, float* __restrict__ maxMaps) {
  __shared__ __align__(16) unsigned short XT[64][40];
  __shared__ float pmax[4][64], psum[4][64];
  int widx = blockIdx.y, b = blockIdx.z;
  int pixBase = blockIdx.x * 64;
  v8f acc[4][2];
#pragma unroll
  for (int i = 0; i < 4; ++i)
#pragma unroll
    for (int j = 0; j < 2; ++j)
#pragma unroll
      for (int e = 0; e < 8; ++e) acc[i][j][e] = 0.f;
  gemm_core(Xp + (size_t)b * C_ * N2_, N2_, pixBase, W3 + (size_t)widx * 65536, XT, acc);
  int tid = threadIdx.x, lane = tid & 31, wave = tid >> 5;
  int waveM = wave >> 1, waveN = wave & 1, half = lane >> 4, l15 = lane & 15;
#pragma unroll
  for (int nt = 0; nt < 2; ++nt) {
    float mx = -1e30f, sm = 0.f;
#pragma unroll
    for (int mt = 0; mt < 4; ++mt)
#pragma unroll
      for (int v = 0; v < 8; ++v) {
        float x = acc[mt][nt][v];
        mx = fmaxf(mx, x);
        sm += x;
      }
    mx = fmaxf(mx, __shfl_xor(mx, 16));
    sm += __shfl_xor(sm, 16);
    if (half == 0) {
      int nl = waveN * 32 + nt * 16 + l15;
      pmax[waveM][nl] = mx;
      psum[waveM][nl] = sm;
    }
  }
  __syncthreads();
  if (tid < 64) {
    float mx = fmaxf(fmaxf(pmax[0][tid], pmax[1][tid]), fmaxf(pmax[2][tid], pmax[3][tid]));
    float sm = psum[0][tid] + psum[1][tid] + psum[2][tid] + psum[3][tid];
    size_t o = (size_t)(widx * B_ + b) * N2_ + pixBase + tid;
    meanMaps[o] = sm * (1.0f / 256.0f);
    maxMaps[o] = mx;
  }
}

// ---------------------------------------------------------------- K7: 7x7 conv on [mean,max] maps -> f,g,h
__global__ void k7_conv7(const float* __restrict__ meanMaps, const float* __restrict__ maxMaps,
                         const float* __restrict__ skw, const float* __restrict__ sqw,
                         const float* __restrict__ svw, float* __restrict__ fgh) {
  __shared__ float wt[98];
  int widx = blockIdx.y, b = blockIdx.z, tid = threadIdx.x;
  if (tid < 98) {
    const float* src = (widx == 0) ? skw : ((widx == 1) ? sqw : svw);
    wt[tid] = src[tid];
  }
  __syncthreads();
  int j = blockIdx.x * 256 + tid;
  int y = j >> 6, x = j & 63;
  const float* mm = meanMaps + (size_t)(widx * B_ + b) * N2_;
  const float* xm = maxMaps + (size_t)(widx * B_ + b) * N2_;
  float s = 0.f;
  for (int dy = 0; dy < 7; ++dy) {
    int yy = y + dy - 3;
    if (yy < 0 || yy >= 64) continue;
    for (int dx = 0; dx < 7; ++dx) {
      int xx = x + dx - 3;
      if (xx < 0 || xx >= 64) continue;
      s += mm[yy * 64 + xx] * wt[dy * 7 + dx] + xm[yy * 64 + xx] * wt[49 + dy * 7 + dx];
    }
  }
  fgh[(size_t)(widx * B_ + b) * N2_ + j] = s;
}

// ---------------------------------------------------------------- K8: rank-1 softmax attention (never materialize NxN)
__global__ void k8_spatial_attn(const float* __restrict__ fgh, float* __restrict__ vS) {
  __shared__ float fL[N2_], hL[N2_], rmax[256], rmin[256];
  int b = blockIdx.y, tid = threadIdx.x;
  const float* f = fgh + (size_t)(0 * B_ + b) * N2_;
  const float* g = fgh + (size_t)(1 * B_ + b) * N2_;
  const float* h = fgh + (size_t)(2 * B_ + b) * N2_;
  float lmax = -1e30f, lmin = 1e30f;
  for (int r = 0; r < 16; ++r) {
    int i = tid + r * 256;
    float fv = f[i];
    fL[i] = fv;
    hL[i] = h[i];
    lmax = fmaxf(lmax, fv);
    lmin = fminf(lmin, fv);
  }
  rmax[tid] = lmax; rmin[tid] = lmin;
  for (int off = 128; off > 0; off >>= 1) {
    __syncthreads();
    if (tid < off) {
      rmax[tid] = fmaxf(rmax[tid], rmax[tid + off]);
      rmin[tid] = fminf(rmin[tid], rmin[tid + off]);
    }
  }
  __syncthreads();
  float fmx = rmax[0], fmn = rmin[0];
  int j = blockIdx.x * 256 + tid;
  float gj = g[j];
  float mj = (gj >= 0.f) ? gj * fmx : gj * fmn;
  float num = 0.f, den = 0.f;
#pragma unroll 4
  for (int i = 0; i < N2_; ++i) {
    float e = __expf(fL[i] * gj - mj);
    den += e;
    num += hL[i] * e;
  }
  vS[(size_t)b * N2_ + j] = num / den;
}

// ---------------------------------------------------------------- K9: bilinear upsample (half-pixel, clamp) + epilogue
__global__ void k9_final(const float* __restrict__ oriS, const float* __restrict__ vS,
                         const float* __restrict__ saw, const float* __restrict__ sab,
                         const float* __restrict__ gs, float* __restrict__ out) {
  size_t idx = (size_t)blockIdx.x * 256 + threadIdx.x;
  int x = (int)(idx & 127);
  int y = (int)((idx >> 7) & 127);
  int c = (int)((idx >> 14) & 255);
  int b = (int)(idx >> 22);
  const float* vb = vS + (size_t)b * N2_;
  float sy = 0.5f * y - 0.25f;
  float sx = 0.5f * x - 0.25f;
  int y0 = (int)floorf(sy), x0 = (int)floorf(sx);
  float wy = sy - (float)y0, wx = sx - (float)x0;
  int y0c = min(63, max(0, y0)), y1c = min(63, max(0, y0 + 1));
  int x0c = min(63, max(0, x0)), x1c = min(63, max(0, x0 + 1));
  float v00 = vb[y0c * 64 + x0c], v01 = vb[y0c * 64 + x1c];
  float v10 = vb[y1c * 64 + x0c], v11 = vb[y1c * 64 + x1c];
  float rv = (1.f - wy) * ((1.f - wx) * v00 + wx * v01) + wy * ((1.f - wx) * v10 + wx * v11);
  out[idx] = gs[0] * (saw[c] * rv + sab[c]) + oriS[idx];
}

// ---------------------------------------------------------------- launch
extern "C" void kernel_launch(void* const* d_in, const int* in_sizes, int n_in,
                              void* d_out, int out_size, void* d_ws, size_t ws_size,
                              hipStream_t stream) {
  (void)in_sizes; (void)n_in; (void)out_size; (void)ws_size;
  const float* t    = (const float*)d_in[0];
  const float* bI   = (const float*)d_in[1];
  const float* m    = (const float*)d_in[2];
  const float* ckw1 = (const float*)d_in[3];
  const float* ckw2 = (const float*)d_in[4];
  const float* cqw1 = (const float*)d_in[5];
  const float* cqw2 = (const float*)d_in[6];
  const float* cvw1 = (const float*)d_in[7];
  const float* cvw2 = (const float*)d_in[8];
  const float* gammac = (const float*)d_in[9];
  const float* ntw  = (const float*)d_in[10];
  const float* nbw  = (const float*)d_in[11];
  const float* nmw  = (const float*)d_in[12];
  const float* skw  = (const float*)d_in[13];
  const float* sqw  = (const float*)d_in[14];
  const float* svw  = (const float*)d_in[15];
  const float* saw  = (const float*)d_in[16];
  const float* sab  = (const float*)d_in[17];
  const float* gammas = (const float*)d_in[18];
  float* out = (float*)d_out;

  char* ws = (char*)d_ws;
  float* ori           = (float*)ws;           ws += 67108864;  // reused as ori_s after K3
  unsigned short* X    = (unsigned short*)ws;  ws += 33554432;
  unsigned short* Xp   = (unsigned short*)ws;  ws += 8388608;
  unsigned short* Wsum = (unsigned short*)ws;  ws += 131072;
  unsigned short* W3   = (unsigned short*)ws;  ws += 393216;
  float* sMean         = (float*)ws;           ws += 12288;
  float* sMax          = (float*)ws;           ws += 12288;
  float* addC          = (float*)ws;           ws += 4096;
  float* meanMaps      = (float*)ws;           ws += 196608;
  float* maxMaps       = (float*)ws;           ws += 196608;
  float* fgh           = (float*)ws;           ws += 196608;
  float* vS            = (float*)ws;           ws += 65536;

  k1_stats_ori<<<1024, 256, 0, stream>>>(t, bI, m, ori, sMean, sMax);
  k2_chan_attn<<<B_, 256, 0, stream>>>(sMean, sMax, ckw1, ckw2, cqw1, cqw2, cvw1, cvw2, gammac, addC);
  k4_wprep<<<256, 256, 0, stream>>>(ntw, nbw, nmw, W3, Wsum);
  k3_fuse_pool<<<16384, 256, 0, stream>>>(ori, addC, X, Xp);
  k5_gemm_full<<<dim3(256, 1, B_), 256, 0, stream>>>(X, Wsum, ori);   // ori buffer now holds ori_s
  k6_gemm_pool<<<dim3(64, 3, B_), 256, 0, stream>>>(Xp, W3, meanMaps, maxMaps);
  k7_conv7<<<dim3(16, 3, B_), 256, 0, stream>>>(meanMaps, maxMaps, skw, sqw, svw, fgh);
  k8_spatial_attn<<<dim3(16, B_), 256, 0, stream>>>(fgh, vS);
  k9_final<<<65536, 256, 0, stream>>>(ori, vS, saw, sab, gammas, out);
}
